// RNN_84988812853300
// MI455X (gfx1250) — matrix operand order repaired and checked by reading, not compiled, not run on hardware
//
#include <hip/hip_runtime.h>
#include <hip/hip_bf16.h>
#include <stdint.h>

// Problem constants (match reference)
#define HH    14
#define WW    14
#define NPIX  196
#define CH    512      // C_IN == C_H == C_OUT == 512
#define BB    2
#define NSEQ  8        // 4 modes * 2 batches
#define NCOL  392      // BB * NPIX
#define NCOLP 400      // padded to multiple of 16 for WMMA N tiles

typedef __attribute__((ext_vector_type(16))) __bf16 v16bf;
typedef __attribute__((ext_vector_type(8)))  float  v8f;

__device__ __forceinline__ unsigned short f2bf(float x) {
  union { float f; unsigned u; } v; v.f = x;
  unsigned u = v.u;
  unsigned r = (u + 0x7FFFu + ((u >> 16) & 1u)) >> 16;  // round-to-nearest-even
  return (unsigned short)r;
}

// ---------------- prep kernels ----------------

__global__ void convert_bf16_kernel(const float* __restrict__ src,
                                    unsigned short* __restrict__ dst, int n) {
  for (int i = blockIdx.x * blockDim.x + threadIdx.x; i < n; i += gridDim.x * blockDim.x)
    dst[i] = f2bf(src[i]);
}

__global__ void zero_f32_kernel(float* __restrict__ p, int n) {
  for (int i = blockIdx.x * blockDim.x + threadIdx.x; i < n; i += gridDim.x * blockDim.x)
    p[i] = 0.0f;
}

// Wh_t[k*CH + c] = Wh[c*CH + k]
__global__ void transpose_kernel(const float* __restrict__ Wh, float* __restrict__ Wht) {
  int i = blockIdx.x * blockDim.x + threadIdx.x;
  if (i < CH * CH) {
    int k = i / CH, c = i % CH;
    Wht[(size_t)k * CH + c] = Wh[(size_t)c * CH + k];
  }
}

// Xt[col*CH + k] = bf16(input[b, k, p]), col = b*NPIX + p; pad cols -> 0
__global__ void build_xt_kernel(const float* __restrict__ input,
                                unsigned short* __restrict__ Xt) {
  int i = blockIdx.x * blockDim.x + threadIdx.x;
  if (i >= NCOLP * CH) return;
  int col = i / CH, k = i % CH;
  unsigned short v = 0;
  if (col < NCOL) {
    int b = col / NPIX, p = col % NPIX;
    v = f2bf(input[((size_t)b * CH + k) * NPIX + p]);
  }
  Xt[(size_t)col * CH + k] = v;
}

// hsum_bf[col*CH + c] = bf16( sum_m hbuf[m][b][p][c] ); pad cols -> 0
__global__ void hsum_bf16_kernel(const float* __restrict__ hbuf,
                                 unsigned short* __restrict__ hs) {
  int i = blockIdx.x * blockDim.x + threadIdx.x;
  if (i >= NCOLP * CH) return;
  int col = i / CH, c = i % CH;
  unsigned short v = 0;
  if (col < NCOL) {
    int b = col / NPIX, p = col % NPIX;
    float s = 0.0f;
    #pragma unroll
    for (int m = 0; m < 4; ++m)
      s += hbuf[(((size_t)(m * BB + b) * NPIX) + p) * CH + c];
    v = f2bf(s);
  }
  hs[(size_t)col * CH + c] = v;
}

// ---------------- WMMA GEMM (bf16 in, f32 acc) ----------------
// D[col*CH + row] = sum_k A[row,k]*B[col,k] + bias_scale*bias[row]
// A: row-major MxK bf16 (M=K=512). B: column-major [col][k] bf16 (NCOLP cols).
__global__ __launch_bounds__(32)
void wmma_gemm_kernel(const unsigned short* __restrict__ Abf,
                      const unsigned short* __restrict__ Bbf,
                      const float* __restrict__ bias, float bias_scale,
                      float* __restrict__ Dout) {
  const int m0 = blockIdx.x * 16;
  const int n0 = blockIdx.y * 16;
  const int lane = threadIdx.x;
  const int l = lane & 15;
  const int h = lane >> 4;

  v8f acc = {0.f, 0.f, 0.f, 0.f, 0.f, 0.f, 0.f, 0.f};

  const uint4* arow = reinterpret_cast<const uint4*>(Abf + (size_t)(m0 + l) * CH);
  const uint4* bcol = reinterpret_cast<const uint4*>(Bbf + (size_t)(n0 + l) * CH);

  for (int kk = 0; kk < CH; kk += 32) {
    union { uint4 u[2]; v16bf v; } af, bf2;
    // A 16x32 layout: lane(l,h): elems 0..7 = K[kk+8h ..], elems 8..15 = K[kk+16+8h ..]
    af.u[0] = arow[(kk + 8 * h) >> 3];
    af.u[1] = arow[(kk + 16 + 8 * h) >> 3];
    // B 32x16 layout: lane(l,h): elems 0..7 = K[kk+8h ..], elems 8..15 = K[kk+16+8h ..] of column n0+l
    bf2.u[0] = bcol[(kk + 8 * h) >> 3];
    bf2.u[1] = bcol[(kk + 16 + 8 * h) >> 3];
    acc = __builtin_amdgcn_wmma_f32_16x16x32_bf16(
        false, af.v, false, bf2.v, (short)0, acc, false, false);
  }

  const int col = n0 + l;
  #pragma unroll
  for (int r = 0; r < 8; ++r) {
    int row = m0 + 8 * h + r;
    Dout[(size_t)col * CH + row] = acc[r] + bias_scale * bias[row];
  }
}

// ---------------- sequential 2-D RNN scan ----------------
// One block per (mode, batch). 512 threads = 16 wave32s.
// g[q][c] = (Wh @ h_vi(q))[c], maintained incrementally in global scratch.
__global__ __launch_bounds__(512)
void scan_kernel(const float* __restrict__ A0,   // [col=(b,p_src)][c] f32 (NCOLP cols)
                 const float* __restrict__ Wht,  // [k][c] f32
                 float* __restrict__ g_ws,       // [seq][q][c] f32, pre-zeroed
                 float* __restrict__ hbuf) {     // [seq][p][c] f32
  const int seq = blockIdx.x;
  const int m = seq >> 1;
  const int b = seq & 1;
  float* g = g_ws + (size_t)seq * NPIX * CH;

  const int tid  = threadIdx.x;
  const int wave = tid >> 5;
  const int lane = tid & 31;

  __shared__ float a_p[CH];
  __shared__ float h_vi[CH];
  __shared__ float part[16][CH];

  for (int p = 0; p < NPIX; ++p) {
    const int px = p / WW, py = p % WW;
    const int fx = (m & 2) ? (HH - 1 - px) : px;
    const int fy = (m & 1) ? (WW - 1 - py) : py;
    const int psrc = fx * WW + fy;

    a_p[tid] = A0[((size_t)b * NPIX + psrc) * CH + tid];
    __syncthreads();

    // accumulate h_vi partials over the causal rectangle
    float accv[16];
    #pragma unroll
    for (int j = 0; j < 16; ++j) accv[j] = 0.0f;

    const int rw = py + 1;
    const int npx = (px + 1) * rw;
    for (int qi = wave; qi < npx; qi += 16) {
      const int qx = qi / rw, qy = qi % rw;
      const float* gq = g + (size_t)(qx * WW + qy) * CH;

      float z[16];
      float mx = 0.0f;  // relu output >= 0
      #pragma unroll
      for (int j = 0; j < 16; ++j) {
        const int c = lane + 32 * j;
        z[j] = fmaxf(a_p[c] + gq[c], 0.0f);
        mx = fmaxf(mx, z[j]);
      }
      #pragma unroll
      for (int off = 16; off > 0; off >>= 1)
        mx = fmaxf(mx, __shfl_xor(mx, off, 32));

      float e[16];
      float s = 0.0f;
      #pragma unroll
      for (int j = 0; j < 16; ++j) { e[j] = __expf(z[j] - mx); s += e[j]; }
      #pragma unroll
      for (int off = 16; off > 0; off >>= 1)
        s += __shfl_xor(s, off, 32);

      const float rs = 1.0f / s;
      #pragma unroll
      for (int j = 0; j < 16; ++j) accv[j] += z[j] * e[j] * rs;
    }

    #pragma unroll
    for (int j = 0; j < 16; ++j) part[wave][lane + 32 * j] = accv[j];
    __syncthreads();

    // reduce 16 waves -> h_vi
    float hv = 0.0f;
    #pragma unroll
    for (int w = 0; w < 16; ++w) hv += part[w][tid];
    h_vi[tid] = hv;
    hbuf[((size_t)seq * NPIX + p) * CH + tid] = hv;
    __syncthreads();

    // g_new[c] = sum_k Wh[c,k] * h_vi[k]  (Wht column-major -> coalesced)
    float gv = 0.0f;
    #pragma unroll 8
    for (int k = 0; k < CH; ++k)
      gv = fmaf(Wht[(size_t)k * CH + tid], h_vi[k], gv);
    g[(size_t)p * CH + tid] = gv;

    __threadfence();   // make g visible to the whole block
    __syncthreads();
  }
}

// ---------------- final channel softmax ----------------
// out[b, c, p] = softmax_c( ypre[col][c] ), col = b*NPIX + p
__global__ __launch_bounds__(512)
void softmax_out_kernel(const float* __restrict__ ypre, float* __restrict__ out) {
  const int col = blockIdx.x;
  const int b = col / NPIX, p = col % NPIX;
  const int c = threadIdx.x;

  __shared__ float red[CH];
  float v = ypre[(size_t)col * CH + c];
  red[c] = v;
  __syncthreads();
  for (int s = 256; s > 0; s >>= 1) {
    if (c < s) red[c] = fmaxf(red[c], red[c + s]);
    __syncthreads();
  }
  const float mx = red[0];
  __syncthreads();
  const float e = __expf(v - mx);
  red[c] = e;
  __syncthreads();
  for (int s = 256; s > 0; s >>= 1) {
    if (c < s) red[c] += red[c + s];
    __syncthreads();
  }
  const float denom = red[0];
  out[((size_t)b * CH + c) * NPIX + p] = e / denom;
}

// ---------------- launcher ----------------
extern "C" void kernel_launch(void* const* d_in, const int* in_sizes, int n_in,
                              void* d_out, int out_size, void* d_ws, size_t ws_size,
                              hipStream_t stream) {
  const float* input = (const float*)d_in[0];  // (B, C_IN, H, W)
  const float* Wx    = (const float*)d_in[1];  // (C_H, C_IN)
  const float* Wh    = (const float*)d_in[2];  // (C_H, C_H)
  const float* b_in  = (const float*)d_in[3];  // (C_H,)
  const float* Wo    = (const float*)d_in[4];  // (C_OUT, C_H)
  const float* b_out = (const float*)d_in[5];  // (C_OUT,)
  float* out = (float*)d_out;

  // workspace layout (256B aligned slices), ~11 MB total
  char* ws = (char*)d_ws;
  size_t off = 0;
  auto take = [&](size_t bytes) { char* p = ws + off; off = (off + bytes + 255) & ~(size_t)255; return p; };
  float*          g_ws  = (float*)         take((size_t)NSEQ * NPIX * CH * 4);
  float*          A0    = (float*)         take((size_t)NCOLP * CH * 4);
  float*          hbuf  = (float*)         take((size_t)NSEQ * NPIX * CH * 4);
  float*          ypre  = (float*)         take((size_t)NCOLP * CH * 4);
  float*          Wht   = (float*)         take((size_t)CH * CH * 4);
  unsigned short* Wx_bf = (unsigned short*)take((size_t)CH * CH * 2);
  unsigned short* Wo_bf = (unsigned short*)take((size_t)CH * CH * 2);
  unsigned short* Xt_bf = (unsigned short*)take((size_t)NCOLP * CH * 2);
  unsigned short* Hs_bf = (unsigned short*)take((size_t)NCOLP * CH * 2);
  (void)ws_size;

  const int T = 256;
  // prep
  zero_f32_kernel<<<(NSEQ * NPIX * CH + T - 1) / T, T, 0, stream>>>(g_ws, NSEQ * NPIX * CH);
  convert_bf16_kernel<<<(CH * CH + T - 1) / T, T, 0, stream>>>(Wx, Wx_bf, CH * CH);
  convert_bf16_kernel<<<(CH * CH + T - 1) / T, T, 0, stream>>>(Wo, Wo_bf, CH * CH);
  transpose_kernel<<<(CH * CH + T - 1) / T, T, 0, stream>>>(Wh, Wht);
  build_xt_kernel<<<(NCOLP * CH + T - 1) / T, T, 0, stream>>>(input, Xt_bf);

  // GEMM1: a = Wx @ x + b_in  (WMMA bf16, M=512, K=512, N=400)
  dim3 gg(CH / 16, NCOLP / 16);
  wmma_gemm_kernel<<<gg, 32, 0, stream>>>(Wx_bf, Xt_bf, b_in, 1.0f, A0);

  // sequential scans: 8 independent (mode, batch) sequences
  scan_kernel<<<NSEQ, 512, 0, stream>>>(A0, Wht, g_ws, hbuf);

  // hsum = sum over modes, to bf16
  hsum_bf16_kernel<<<(NCOLP * CH + T - 1) / T, T, 0, stream>>>(hbuf, Hs_bf);

  // GEMM2: y = Wo @ hsum + 4*b_out  (WMMA bf16)
  wmma_gemm_kernel<<<gg, 32, 0, stream>>>(Wo_bf, Hs_bf, b_out, 4.0f, ypre);

  // final channel softmax -> d_out (B, C_OUT, H, W)
  softmax_out_kernel<<<NCOL, CH, 0, stream>>>(ypre, out);
}
